// VectorQuantizer_87952340288148
// MI455X (gfx1250) — matrix operand order repaired and checked by reading, not compiled, not run on hardware
//
#include <hip/hip_runtime.h>

// ---------------------------------------------------------------------------
// SimVQ vector quantizer for MI455X (gfx1250, wave32).
// Dominant cost: 16384x8192x256 distance GEMM (68.7 GFLOP) fused with
// row-argmin on v_wmma_f32_16x16x32_bf16. mean_distance computed analytically
// from factored sums; the NxK matrix is never materialized.
// ---------------------------------------------------------------------------

typedef __attribute__((ext_vector_type(16))) __bf16 v16bf;
typedef __attribute__((ext_vector_type(8)))  __bf16 v8bf;
typedef __attribute__((ext_vector_type(8)))  float  v8f;

#define N_ROWS 16384   // B*T*H*W
#define K_CB   8192
#define D_DIM  256
#define SPAT   8192    // T*H*W per batch

// ---- fragment loaders (ISA 7.12.2 layouts) --------------------------------
// A 16x32 bf16: lanes 0-15 hold K {0..7,16..23}, lanes 16-31 hold {8..15,24..31}
__device__ __forceinline__ v16bf load_afrag(const __bf16* __restrict__ rowptr,
                                            int kbase, int half) {
  v8bf lo = *(const v8bf*)(rowptr + kbase + 8 * half);
  v8bf hi = *(const v8bf*)(rowptr + kbase + 16 + 8 * half);
  return __builtin_shufflevector(lo, hi, 0,1,2,3,4,5,6,7,8,9,10,11,12,13,14,15);
}
// B 32x16 bf16: lanes 0-15 hold K {0..7,16..23}, lanes 16-31 hold {8..15,24..31}
__device__ __forceinline__ v16bf load_bfrag(const __bf16* __restrict__ colrow,
                                            int kbase, int half) {
  v8bf lo = *(const v8bf*)(colrow + kbase + 8 * half);
  v8bf hi = *(const v8bf*)(colrow + kbase + 16 + 8 * half);
  return __builtin_shufflevector(lo, hi, 0,1,2,3,4,5,6,7,8,9,10,11,12,13,14,15);
}

__device__ __forceinline__ v8f wmma_bf16(v16bf a, v16bf b, v8f c) {
  return __builtin_amdgcn_wmma_f32_16x16x32_bf16(false, a, false, b,
                                                 (short)0, c, false, false);
}

// ---- fp32 -> bf16 convert --------------------------------------------------
__global__ __launch_bounds__(256) void cvt_bf16(const float* __restrict__ src,
                                                __bf16* __restrict__ dst, int n) {
  int i = blockIdx.x * 256 + threadIdx.x;
  if (i < n) dst[i] = (__bf16)src[i];
}

// ---- z [B,D,S] -> z_flat [B*S, D] (fp32 + bf16), LDS tile transpose --------
__global__ __launch_bounds__(256) void transpose_z(const float* __restrict__ z,
                                                   float* __restrict__ zf,
                                                   __bf16* __restrict__ zb) {
  __shared__ float t[32][33];
  const int b = blockIdx.z, d0 = blockIdx.y * 32, s0 = blockIdx.x * 32;
  const int tid = threadIdx.x;
  {
    const int sl = tid & 31, dl0 = tid >> 5;
    const float* src = z + ((size_t)b * D_DIM + d0) * SPAT + s0;
#pragma unroll
    for (int q = 0; q < 4; ++q) {
      int dl = dl0 + q * 8;
      t[dl][sl] = src[(size_t)dl * SPAT + sl];
    }
  }
  __syncthreads();
  {
    const int dl = tid & 31, sl0 = tid >> 5;
#pragma unroll
    for (int q = 0; q < 4; ++q) {
      int sl = sl0 + q * 8;
      float v = t[dl][sl];
      size_t r = (size_t)b * SPAT + s0 + sl;
      zf[r * D_DIM + d0 + dl] = v;
      zb[r * D_DIM + d0 + dl] = (__bf16)v;
    }
  }
}

// ---- row/col statistics: sumvec[d] += col sums, sumsq += sum of row norms,
//      optional per-row squared norm (needed for codebook) -------------------
__global__ __launch_bounds__(256) void rowstats(const float* __restrict__ src,
                                                int rows_per_block,
                                                float* __restrict__ rowsq,
                                                float* __restrict__ sumvec,
                                                float* __restrict__ sumsq) {
  const int tid = threadIdx.x, lane = tid & 31, wid = tid >> 5;
  const int r0 = blockIdx.x * rows_per_block;
  float colacc[8] = {0.f, 0.f, 0.f, 0.f, 0.f, 0.f, 0.f, 0.f};
  float sqacc = 0.f;
  for (int r = wid; r < rows_per_block; r += 8) {
    const float* row = src + (size_t)(r0 + r) * D_DIM;
    float s = 0.f;
#pragma unroll
    for (int q = 0; q < 8; ++q) {
      float v = row[lane + 32 * q];
      colacc[q] += v;
      s += v * v;
    }
#pragma unroll
    for (int off = 16; off >= 1; off >>= 1) s += __shfl_xor(s, off, 32);
    if (lane == 0) {
      if (rowsq) rowsq[r0 + r] = s;
      sqacc += s;
    }
  }
#pragma unroll
  for (int q = 0; q < 8; ++q) atomicAdd(&sumvec[lane + 32 * q], colacc[q]);
  if (lane == 0) atomicAdd(sumsq, sqacc);
}

// ---- MLP GEMM 1: H = gelu(emb @ w1^T), bf16 out ----------------------------
__global__ __launch_bounds__(256) void mlp_gemm_gelu(const __bf16* __restrict__ A,
                                                     const __bf16* __restrict__ B,
                                                     __bf16* __restrict__ H) {
  const int lane = threadIdx.x & 31, wid = threadIdx.x >> 5;
  const int half = lane >> 4;
  const int arow = blockIdx.x * 128 + wid * 16 + (lane & 15);
  const int colbase = blockIdx.y * 16;
  const __bf16* ap = A + (size_t)arow * D_DIM;
  const __bf16* bp = B + (size_t)(colbase + (lane & 15)) * D_DIM;
  v8f c = {0.f, 0.f, 0.f, 0.f, 0.f, 0.f, 0.f, 0.f};
#pragma unroll
  for (int kk = 0; kk < 8; ++kk)
    c = wmma_bf16(load_afrag(ap, kk * 32, half), load_bfrag(bp, kk * 32, half), c);
  const int n = colbase + (lane & 15);
  const int mbase = blockIdx.x * 128 + wid * 16 + 8 * half;
#pragma unroll
  for (int v = 0; v < 8; ++v) {
    float x = c[v];
    float g = 0.5f * x * (1.0f + erff(x * 0.70710678118654752f));
    H[(size_t)(mbase + v) * D_DIM + n] = (__bf16)g;
  }
}

// ---- MLP GEMM 2: CB = H @ w2^T, fp32 + bf16 out ----------------------------
__global__ __launch_bounds__(256) void mlp_gemm_out(const __bf16* __restrict__ A,
                                                    const __bf16* __restrict__ B,
                                                    float* __restrict__ CB,
                                                    __bf16* __restrict__ CBB) {
  const int lane = threadIdx.x & 31, wid = threadIdx.x >> 5;
  const int half = lane >> 4;
  const int arow = blockIdx.x * 128 + wid * 16 + (lane & 15);
  const int colbase = blockIdx.y * 16;
  const __bf16* ap = A + (size_t)arow * D_DIM;
  const __bf16* bp = B + (size_t)(colbase + (lane & 15)) * D_DIM;
  v8f c = {0.f, 0.f, 0.f, 0.f, 0.f, 0.f, 0.f, 0.f};
#pragma unroll
  for (int kk = 0; kk < 8; ++kk)
    c = wmma_bf16(load_afrag(ap, kk * 32, half), load_bfrag(bp, kk * 32, half), c);
  const int n = colbase + (lane & 15);
  const int mbase = blockIdx.x * 128 + wid * 16 + 8 * half;
#pragma unroll
  for (int v = 0; v < 8; ++v) {
    float x = c[v];
    CB[(size_t)(mbase + v) * D_DIM + n] = x;
    CBB[(size_t)(mbase + v) * D_DIM + n] = (__bf16)x;
  }
}

// ---- fused distance GEMM + row argmin --------------------------------------
// Block: 8 waves x 16 rows = 128 z-rows. A-frags (K=256) persistent in VGPRs.
// Streams 16 codebook columns / iter through double-buffered LDS (8KB/buf).
// d' = ||cb||^2 - 2*z.cb  (||z||^2 constant per row -> dropped for argmin).
__global__ __launch_bounds__(256) void argmin_kernel(const __bf16* __restrict__ zb,
                                                     const __bf16* __restrict__ cbb,
                                                     const float* __restrict__ cbsq,
                                                     int* __restrict__ idx_out) {
  __shared__ __align__(16) __bf16 lds_cb[2][16 * D_DIM];  // 2 x 8 KB
  __shared__ float lds_sq[2][16];
  const int tid = threadIdx.x;
  const int lane = tid & 31, wid = tid >> 5;
  const int half = lane >> 4;
  const int rowbase = blockIdx.x * 128 + wid * 16;
  const int arow = rowbase + (lane & 15);

  // persistent A fragments: 8 K-chunks of 32
  v16bf a[8];
  const __bf16* ap = zb + (size_t)arow * D_DIM;
#pragma unroll
  for (int kk = 0; kk < 8; ++kk) a[kk] = load_afrag(ap, kk * 32, half);

  float runmin[8];
  int runidx[8];
#pragma unroll
  for (int v = 0; v < 8; ++v) { runmin[v] = 3.4e38f; runidx[v] = 0; }

  // cooperative staging: thread t copies 32B of codebook row (t>>4)
  const int srow = tid >> 4;
  const int scol = (tid & 15) * 16;
  uint4 r0 = *(const uint4*)(cbb + (size_t)srow * D_DIM + scol);
  uint4 r1 = *(const uint4*)(cbb + (size_t)srow * D_DIM + scol + 8);
  float rsq = (tid < 16) ? cbsq[tid] : 0.0f;

  const int NJ = K_CB / 16;  // 512 column chunks
  for (int j = 0; j < NJ; ++j) {
    const int buf = j & 1;
    *(uint4*)(&lds_cb[buf][srow * D_DIM + scol]) = r0;
    *(uint4*)(&lds_cb[buf][srow * D_DIM + scol + 8]) = r1;
    if (tid < 16) lds_sq[buf][tid] = rsq;
    __syncthreads();

    if (j + 1 < NJ) {  // register prefetch of next chunk
      const int jb = (j + 1) * 16;
      r0 = *(const uint4*)(cbb + (size_t)(jb + srow) * D_DIM + scol);
      r1 = *(const uint4*)(cbb + (size_t)(jb + srow) * D_DIM + scol + 8);
      if (tid < 16) rsq = cbsq[jb + tid];
    }

    v8f c = {0.f, 0.f, 0.f, 0.f, 0.f, 0.f, 0.f, 0.f};
    const __bf16* bp = &lds_cb[buf][(lane & 15) * D_DIM];
#pragma unroll
    for (int kk = 0; kk < 8; ++kk)
      c = wmma_bf16(a[kk], load_bfrag(bp, kk * 32, half), c);

    const float sq = lds_sq[buf][lane & 15];
    const int col = j * 16 + (lane & 15);
#pragma unroll
    for (int v = 0; v < 8; ++v) {
      float dv = fmaf(-2.0f, c[v], sq);
      if (dv < runmin[v]) { runmin[v] = dv; runidx[v] = col; }
    }
    __syncthreads();
  }

  // reduce (min, argmin) across the 16 lanes of each half-wave
#pragma unroll
  for (int v = 0; v < 8; ++v) {
    float mv = runmin[v];
    int mi = runidx[v];
#pragma unroll
    for (int off = 8; off >= 1; off >>= 1) {
      float om = __shfl_xor(mv, off, 32);
      int oi = __shfl_xor(mi, off, 32);
      if (om < mv || (om == mv && oi < mi)) { mv = om; mi = oi; }
    }
    if ((lane & 15) == 0) idx_out[rowbase + v + 8 * half] = mi;
  }
}

// ---- code-usage histogram ---------------------------------------------------
__global__ __launch_bounds__(256) void count_kernel(const int* __restrict__ idx,
                                                    float* __restrict__ counts, int n) {
  int i = blockIdx.x * 256 + threadIdx.x;
  if (i < n) atomicAdd(&counts[idx[i]], 1.0f);
}

// ---- gather z_q, transposed write-back to [B,D,S], fused loss accumulation --
__global__ __launch_bounds__(256) void gather_kernel(const int* __restrict__ idx,
                                                     const float* __restrict__ cb,
                                                     const float* __restrict__ zf,
                                                     float* __restrict__ out,
                                                     float* __restrict__ loss_acc) {
  __shared__ float tile[32][257];
  __shared__ int sidx[32];
  __shared__ float wsum[8];
  const int b = blockIdx.y;
  const int s0 = blockIdx.x * 32;
  const int tid = threadIdx.x;
  const size_t rbase = (size_t)b * SPAT + s0;
  if (tid < 32) sidx[tid] = idx[rbase + tid];
  __syncthreads();
  float acc = 0.f;
  for (int sl = 0; sl < 32; ++sl) {
    float v = cb[(size_t)sidx[sl] * D_DIM + tid];   // coalesced across d
    tile[sl][tid] = v;
    float dz = v - zf[(rbase + sl) * D_DIM + tid];
    acc += dz * dz;
  }
  __syncthreads();
  {
    const int sl = tid & 31, d0 = tid >> 5;
    float* ob = out + (size_t)b * (D_DIM * SPAT) + s0;
#pragma unroll
    for (int q = 0; q < 32; ++q) {
      int d = d0 + q * 8;
      ob[(size_t)d * SPAT + sl] = tile[sl][d];      // coalesced across s
    }
  }
  const int lane = tid & 31;
#pragma unroll
  for (int off = 16; off >= 1; off >>= 1) acc += __shfl_xor(acc, off, 32);
  if (lane == 0) wsum[tid >> 5] = acc;
  __syncthreads();
  if (tid == 0) {
    float s = 0.f;
    for (int w = 0; w < 8; ++w) s += wsum[w];
    atomicAdd(loss_acc, s);
  }
}

// ---- scalars: loss, perplexity, analytic mean distance ----------------------
__global__ __launch_bounds__(256) void finalize_kernel(const float* __restrict__ counts,
                                                       const float* __restrict__ sumz,
                                                       const float* __restrict__ sumcb,
                                                       const float* __restrict__ scal,
                                                       float* __restrict__ outs) {
  __shared__ float red[256];
  const int tid = threadIdx.x;
  float h = 0.f;
  for (int k = tid; k < K_CB; k += 256) {
    float p = counts[k] * (1.0f / (float)N_ROWS);
    h += p * logf(p + 1e-10f);
  }
  red[tid] = h;
  __syncthreads();
  for (int s = 128; s > 0; s >>= 1) {
    if (tid < s) red[tid] += red[tid + s];
    __syncthreads();
  }
  float hsum = red[0];
  __syncthreads();
  red[tid] = sumz[tid] * sumcb[tid];
  __syncthreads();
  for (int s = 128; s > 0; s >>= 1) {
    if (tid < s) red[tid] += red[tid + s];
    __syncthreads();
  }
  if (tid == 0) {
    float dot = red[0];
    float loss = 1.25f * scal[2] / ((float)N_ROWS * (float)D_DIM);
    float perp = expf(-hsum);
    float md = ((float)K_CB * scal[0] + (float)N_ROWS * scal[1] - 2.0f * dot) /
               ((float)N_ROWS * (float)K_CB);
    outs[0] = loss;
    outs[1] = perp;
    outs[2] = md;
  }
}

// ---------------------------------------------------------------------------
extern "C" void kernel_launch(void* const* d_in, const int* in_sizes, int n_in,
                              void* d_out, int out_size, void* d_ws, size_t ws_size,
                              hipStream_t stream) {
  const float* z   = (const float*)d_in[0];   // [2,256,8,32,32]
  const float* emb = (const float*)d_in[1];   // [8192,256]
  const float* w1  = (const float*)d_in[2];   // [256,256]
  const float* w2  = (const float*)d_in[3];   // [256,256]
  float* out = (float*)d_out;                 // z_q (4194304) + loss,perp,mean_d
  char* ws = (char*)d_ws;

  // accumulator region (zeroed each call)
  float* accf  = (float*)ws;
  float* sumz  = accf;         // [256]
  float* sumcb = accf + 256;   // [256]
  float* scal  = accf + 512;   // [0]=sum||z||^2 [1]=sum||cb||^2 [2]=loss acc
  float* counts = accf + 1024; // [8192]
  size_t off = (size_t)(1024 + K_CB) * 4;  // 36864, 256B aligned

  float*  zf   = (float*)(ws + off);  off += (size_t)N_ROWS * D_DIM * 4;  // 16 MB
  __bf16* zb   = (__bf16*)(ws + off); off += (size_t)N_ROWS * D_DIM * 2;  //  8 MB
  __bf16* hb   = (__bf16*)(ws + off); off += (size_t)K_CB * D_DIM * 2;    //  4 MB
  float*  cb   = (float*)(ws + off);  off += (size_t)K_CB * D_DIM * 4;    //  8 MB
  __bf16* cbb  = (__bf16*)(ws + off); off += (size_t)K_CB * D_DIM * 2;    //  4 MB
  float*  cbsq = (float*)(ws + off);  off += (size_t)K_CB * 4;
  __bf16* embb = (__bf16*)(ws + off); off += (size_t)K_CB * D_DIM * 2;    //  4 MB
  __bf16* w1b  = (__bf16*)(ws + off); off += (size_t)D_DIM * D_DIM * 2;
  __bf16* w2b  = (__bf16*)(ws + off); off += (size_t)D_DIM * D_DIM * 2;
  int*    idx  = (int*)(ws + off);    off += (size_t)N_ROWS * 4;

  hipMemsetAsync(accf, 0, (size_t)(1024 + K_CB) * 4, stream);

  cvt_bf16<<<(K_CB * D_DIM + 255) / 256, 256, 0, stream>>>(emb, embb, K_CB * D_DIM);
  cvt_bf16<<<(D_DIM * D_DIM + 255) / 256, 256, 0, stream>>>(w1, w1b, D_DIM * D_DIM);
  cvt_bf16<<<(D_DIM * D_DIM + 255) / 256, 256, 0, stream>>>(w2, w2b, D_DIM * D_DIM);

  transpose_z<<<dim3(SPAT / 32, D_DIM / 32, 2), 256, 0, stream>>>(z, zf, zb);
  rowstats<<<64, 256, 0, stream>>>(zf, 256, nullptr, sumz, &scal[0]);

  mlp_gemm_gelu<<<dim3(K_CB / 128, D_DIM / 16), 256, 0, stream>>>(embb, w1b, hb);
  mlp_gemm_out<<<dim3(K_CB / 128, D_DIM / 16), 256, 0, stream>>>(hb, w2b, cb, cbb);
  rowstats<<<32, 256, 0, stream>>>(cb, 256, cbsq, sumcb, &scal[1]);

  argmin_kernel<<<N_ROWS / 128, 256, 0, stream>>>(zb, cbb, cbsq, idx);

  count_kernel<<<N_ROWS / 256, 256, 0, stream>>>(idx, counts, N_ROWS);
  gather_kernel<<<dim3(SPAT / 32, 2), 256, 0, stream>>>(idx, cb, zf, out, &scal[2]);
  finalize_kernel<<<1, 256, 0, stream>>>(counts, sumz, sumcb, scal,
                                         out + (size_t)2 * D_DIM * SPAT);
}
